// GraphAttentionLayer_53772990546391
// MI455X (gfx1250) — compile-verified, matrix operand order checked by reading, not run on hardware
//
#include <hip/hip_runtime.h>
#include <hip/hip_bf16.h>

typedef __attribute__((ext_vector_type(16))) _Float16 v16h;
typedef __attribute__((ext_vector_type(8)))  _Float16 v8h;
typedef __attribute__((ext_vector_type(8)))  float    v8f;

#define DEVFN static __device__ __forceinline__

#define N_NODES   8192
#define IN_DIM    256
#define OUT_DIM   128
#define NEG_SLOPE 0.2f

// ---------------- workspace layout (bytes, 256-aligned) ----------------
// Phase 1 (convert/pack/GEMM/proj):           Phase 2 (stats/attn):
//   [0 .. 4.19M)  hh   (h in f16)               [0 .. 8.39M)  bitmask (adj!=0, 1 bit/edge)
//   [4.19M..4.26M) Wp  (packed W frags)         (hh/Wp/Wh-f32 are dead by then)
//   [4.26M..8.45M) Wh  f32
//   [8.45M..10.55M) Whp (packed Wh f16 frags)  -- live through attn
//   s/t/m/l vectors after that                 -- live through attn
#define OFF_HH   (size_t)0
#define OFF_BM   (size_t)0                      // bitmask: 8192*256*4 = 8388608 (overlays phase-1 bufs)
#define OFF_WP   (size_t)4194304
#define OFF_WH   (size_t)4259840
#define OFF_WHP  (size_t)8454144                // 2097152
#define OFF_S    (size_t)10551296
#define OFF_T    (size_t)10584064
#define OFF_M    (size_t)10616832
#define OFF_L    (size_t)10649600

// ---------------- kernel 1: f32 -> f16 convert of h ----------------
__global__ void gat_convert_h(const float* __restrict__ h, _Float16* __restrict__ hh, int n) {
    int i = blockIdx.x * blockDim.x + threadIdx.x;
    if (i < n) hh[i] = (_Float16)h[i];
}

// ---------------- kernel 2: pack a (K x 128) f32 matrix into WMMA B fragments ----
// frag = kt2*8 + nt; per lane 16 contiguous halves:
// n = nt*16 + (lane&15); kb = kt2*32 + (lane>>4)*8; e<8 -> K=kb+e; e>=8 -> K=kb+16+(e-8)
__global__ void gat_pack_b(const float* __restrict__ src, _Float16* __restrict__ dst, int nfrag) {
    int g = blockIdx.x * blockDim.x + threadIdx.x;
    if (g >= nfrag * 32) return;
    int frag = g >> 5;
    int lane = g & 31;
    int kt2  = frag >> 3;
    int nt   = frag & 7;
    int kb   = kt2 * 32 + ((lane >> 4) * 8);
    int n    = nt * 16 + (lane & 15);
    _Float16* d = dst + ((size_t)frag * 32 + lane) * 16;
#pragma unroll
    for (int e = 0; e < 8; ++e) {
        d[e]     = (_Float16)src[(size_t)(kb + e)      * OUT_DIM + n];
        d[e + 8] = (_Float16)src[(size_t)(kb + 16 + e) * OUT_DIM + n];
    }
}

// ---------------- kernel 3: Wh = h @ W via v_wmma_f32_16x16x32_f16 ----------------
__global__ void gat_gemm_wh(const _Float16* __restrict__ hh, const _Float16* __restrict__ wp,
                            float* __restrict__ wh) {
    int tid  = threadIdx.x;
    int lane = tid & 31;
    int nt   = tid >> 5;       // 0..7
    int rt   = blockIdx.x;     // 0..511
    int irow = rt * 16 + (lane & 15);
    v8f acc = {};
#pragma unroll
    for (int kt = 0; kt < 8; ++kt) {
        int kb = kt * 32 + ((lane >> 4) * 8);
        v8h lo = *(const v8h*)(hh + (size_t)irow * IN_DIM + kb);
        v8h hi = *(const v8h*)(hh + (size_t)irow * IN_DIM + kb + 16);
        v16h A;
#pragma unroll
        for (int e = 0; e < 8; ++e) { A[e] = lo[e]; A[e + 8] = hi[e]; }
        v16h B = *(const v16h*)(wp + ((size_t)(kt * 8 + nt) * 32 + lane) * 16);
        acc = __builtin_amdgcn_wmma_f32_16x16x32_f16(false, A, false, B, (short)0, acc, false, false);
    }
    int row0 = rt * 16 + ((lane >= 16) ? 8 : 0);
    int col  = nt * 16 + (lane & 15);
#pragma unroll
    for (int r = 0; r < 8; ++r)
        wh[(size_t)(row0 + r) * OUT_DIM + col] = acc[r];
}

// ---------------- kernel 4: s = Wh @ a_src, t = Wh @ a_dst ----------------
__global__ void gat_proj_st(const float* __restrict__ wh, const float* __restrict__ a,
                            float* __restrict__ s, float* __restrict__ t) {
    int i = blockIdx.x * blockDim.x + threadIdx.x;
    if (i >= N_NODES) return;
    const float4* row  = (const float4*)(wh + (size_t)i * OUT_DIM);
    const float4* asrc = (const float4*)(a);
    const float4* adst = (const float4*)(a + OUT_DIM);
    float ss = 0.f, tt = 0.f;
#pragma unroll 8
    for (int q = 0; q < OUT_DIM / 4; ++q) {
        float4 v = row[q];
        float4 x = asrc[q];
        float4 y = adst[q];
        ss += v.x * x.x + v.y * x.y + v.z * x.z + v.w * x.w;
        tt += v.x * y.x + v.y * y.y + v.z * y.z + v.w * y.w;
    }
    s[i] = ss;
    t[i] = tt;
}

// ---------------- kernel 5: online-softmax row stats + bitmask compaction ------
// one block (256 threads = 8 wave32) per row. The ONLY pass over the 268MB adj.
// Also emits a 1-bit/edge mask via wave32 ballot so the attention pass never
// touches adj again (8.4MB instead of 268MB).
__global__ void gat_row_stats(const int* __restrict__ adj, const float* __restrict__ s,
                              const float* __restrict__ t, float* __restrict__ mrow,
                              float* __restrict__ lrow, unsigned* __restrict__ bm) {
    __shared__ float sm[256];
    __shared__ float sl[256];
    int i    = blockIdx.x;
    int tid  = threadIdx.x;
    int lane = tid & 31;
    float si = s[i];
    const int* arow = adj + (size_t)i * N_NODES;
    unsigned* bmrow = bm + (size_t)i * (N_NODES / 32);
    float m = -__builtin_inff();
    float l = 0.f;
#pragma unroll 4
    for (int jj = 0; jj < N_NODES / 256; ++jj) {
        int j  = jj * 256 + tid;
        int av = arow[j];
        // compact mask: 32 consecutive j per wave -> one word
        unsigned word = __builtin_amdgcn_ballot_w32(av != 0);
        if (lane == 0) bmrow[j >> 5] = word;
        float x = si + t[j];
        float e = x > 0.f ? x : NEG_SLOPE * x;
        if (av != 0) {
            if (e > m) { l *= __expf(m - e); m = e; }
            l += __expf(e - m);
        }
    }
    sm[tid] = m; sl[tid] = l;
    __syncthreads();
    for (int off = 128; off > 0; off >>= 1) {
        if (tid < off) {
            float m1 = sm[tid], l1 = sl[tid];
            float m2 = sm[tid + off], l2 = sl[tid + off];
            float mm = fmaxf(m1, m2);
            float ll = 0.f;
            if (m1 > -__builtin_inff()) ll += l1 * __expf(m1 - mm);
            if (m2 > -__builtin_inff()) ll += l2 * __expf(m2 - mm);
            sm[tid] = mm; sl[tid] = ll;
        }
        __syncthreads();
    }
    if (tid == 0) { mrow[i] = sm[0]; lrow[i] = sl[0]; }
}

DEVFN _Float16 gat_p(unsigned bit, float tj, float si, float mi) {
    float x = si + tj;
    float e = x > 0.f ? x : NEG_SLOPE * x;
    float p = bit ? __expf(e - mi) : 0.f;
    return (_Float16)p;
}

// ---------------- kernel 6: flash-style alpha @ Wh ----------------
// block = 256 threads (8 waves) per 16-row tile; wave w scans j in [w*1024,(w+1)*1024),
// builds the 16x32 f16 P fragment in registers from the 1-bit mask, runs 8 WMMAs
// (all 128 output cols), then LDS-reduces across waves and normalizes by l_i.
__global__ void gat_attn(const unsigned* __restrict__ bm, const _Float16* __restrict__ whp,
                         const float* __restrict__ s, const float* __restrict__ t,
                         const float* __restrict__ mrow, const float* __restrict__ lrow,
                         float* __restrict__ out) {
    __shared__ float red[16 * OUT_DIM];
    int tid  = threadIdx.x;
    int lane = tid & 31;
    int wave = tid >> 5;     // 0..7
    int rt   = blockIdx.x;   // 0..511

    int   ilane = rt * 16 + (lane & 15);
    float si    = s[ilane];
    float mi    = mrow[ilane];
    const unsigned* bmrow = bm + (size_t)ilane * (N_NODES / 32);
    int kb = (lane >> 4) * 8;

    v8f acc[8];
#pragma unroll
    for (int q = 0; q < 8; ++q) acc[q] = (v8f){};

    for (int kk = 0; kk < 32; ++kk) {
        int j0 = wave * 1024 + kk * 32;
        int jA = j0 + kb;
        int jB = jA + 16;
        __builtin_prefetch(bmrow + (j0 >> 5) + 8, 0, 1);  // next mask words
        unsigned w32 = bmrow[j0 >> 5];                    // bits for all 32 j of this step
        float4 t0 = *(const float4*)(t + jA);
        float4 t1 = *(const float4*)(t + jA + 4);
        float4 t2 = *(const float4*)(t + jB);
        float4 t3 = *(const float4*)(t + jB + 4);
        v16h A;
        A[0]  = gat_p((w32 >> (kb + 0))  & 1u, t0.x, si, mi);
        A[1]  = gat_p((w32 >> (kb + 1))  & 1u, t0.y, si, mi);
        A[2]  = gat_p((w32 >> (kb + 2))  & 1u, t0.z, si, mi);
        A[3]  = gat_p((w32 >> (kb + 3))  & 1u, t0.w, si, mi);
        A[4]  = gat_p((w32 >> (kb + 4))  & 1u, t1.x, si, mi);
        A[5]  = gat_p((w32 >> (kb + 5))  & 1u, t1.y, si, mi);
        A[6]  = gat_p((w32 >> (kb + 6))  & 1u, t1.z, si, mi);
        A[7]  = gat_p((w32 >> (kb + 7))  & 1u, t1.w, si, mi);
        A[8]  = gat_p((w32 >> (kb + 16)) & 1u, t2.x, si, mi);
        A[9]  = gat_p((w32 >> (kb + 17)) & 1u, t2.y, si, mi);
        A[10] = gat_p((w32 >> (kb + 18)) & 1u, t2.z, si, mi);
        A[11] = gat_p((w32 >> (kb + 19)) & 1u, t2.w, si, mi);
        A[12] = gat_p((w32 >> (kb + 20)) & 1u, t3.x, si, mi);
        A[13] = gat_p((w32 >> (kb + 21)) & 1u, t3.y, si, mi);
        A[14] = gat_p((w32 >> (kb + 22)) & 1u, t3.z, si, mi);
        A[15] = gat_p((w32 >> (kb + 23)) & 1u, t3.w, si, mi);
        int jt2 = j0 >> 5;
#pragma unroll
        for (int nt = 0; nt < 8; ++nt) {
            v16h B = *(const v16h*)(whp + ((size_t)(jt2 * 8 + nt) * 32 + lane) * 16);
            acc[nt] = __builtin_amdgcn_wmma_f32_16x16x32_f16(false, A, false, B, (short)0,
                                                             acc[nt], false, false);
        }
    }

    // cross-wave reduction in LDS
    for (int q = tid; q < 16 * OUT_DIM; q += 256) red[q] = 0.f;
    __syncthreads();
    int rbase = (lane >= 16) ? 8 : 0;
    int coll  = lane & 15;
#pragma unroll
    for (int nt = 0; nt < 8; ++nt) {
#pragma unroll
        for (int r = 0; r < 8; ++r)
            atomicAdd(&red[(r + rbase) * OUT_DIM + nt * 16 + coll], acc[nt][r]);
    }
    __syncthreads();
    for (int q = tid; q < 16 * OUT_DIM; q += 256) {
        int rr  = q >> 7;
        int cc  = q & 127;
        int row = rt * 16 + rr;
        float lv = lrow[row];
        out[(size_t)row * OUT_DIM + cc] = (lv > 0.f) ? red[q] / lv : 0.f;
    }
}

extern "C" void kernel_launch(void* const* d_in, const int* in_sizes, int n_in,
                              void* d_out, int out_size, void* d_ws, size_t ws_size,
                              hipStream_t stream) {
    (void)in_sizes; (void)n_in; (void)out_size; (void)ws_size;
    const float* h   = (const float*)d_in[0];
    const int*   adj = (const int*)d_in[1];
    const float* W   = (const float*)d_in[2];
    const float* a   = (const float*)d_in[3];
    float* out = (float*)d_out;

    char* ws = (char*)d_ws;
    _Float16* hh  = (_Float16*)(ws + OFF_HH);
    unsigned* bm  = (unsigned*)(ws + OFF_BM);   // overlays hh/Wp/Wh-f32 (dead by stats)
    _Float16* wp  = (_Float16*)(ws + OFF_WP);
    float*    wh  = (float*)   (ws + OFF_WH);
    _Float16* whp = (_Float16*)(ws + OFF_WHP);
    float*    s   = (float*)   (ws + OFF_S);
    float*    t   = (float*)   (ws + OFF_T);
    float*    m   = (float*)   (ws + OFF_M);
    float*    l   = (float*)   (ws + OFF_L);

    // 1) h -> f16
    gat_convert_h<<<(N_NODES * IN_DIM + 255) / 256, 256, 0, stream>>>(h, hh, N_NODES * IN_DIM);
    // 2) pack W (64 fragments)
    gat_pack_b<<<(64 * 32 + 255) / 256, 256, 0, stream>>>(W, wp, 64);
    // 3) Wh = h @ W
    gat_gemm_wh<<<N_NODES / 16, 256, 0, stream>>>(hh, wp, wh);
    // 4) pack Wh as B fragments (2048 fragments)
    gat_pack_b<<<(2048 * 32 + 255) / 256, 256, 0, stream>>>(wh, whp, 2048);
    // 5) s, t projections
    gat_proj_st<<<N_NODES / 256, 256, 0, stream>>>(wh, a, s, t);
    // 6) softmax row stats + bitmask compaction (single pass over adj)
    gat_row_stats<<<N_NODES, 256, 0, stream>>>(adj, s, t, m, l, bm);
    // 7) flash-style attention output (reads 8.4MB bitmask, not 268MB adj)
    gat_attn<<<N_NODES / 16, 256, 0, stream>>>(bm, whp, s, t, m, l, out);
}
